// WindowAttention_69054484185172
// MI455X (gfx1250) — matrix operand order repaired and checked by reading, not compiled, not run on hardware
//
#include <hip/hip_runtime.h>
#include <hip/hip_bf16.h>

typedef __bf16 bf16;
typedef __attribute__((ext_vector_type(16))) __bf16 v16bf;
typedef __attribute__((ext_vector_type(8)))  __bf16 v8bf;
typedef __attribute__((ext_vector_type(8)))  float  v8f;

#define NT     256   // tokens per window
#define HEADS  6
#define DIM    192
#define HD     32
#define BWIN   256   // B_
#define NWMASK 64
#define TBL    961   // (2*16-1)^2

// ---------------------------------------------------------------------------
// CDNA5 helpers
// ---------------------------------------------------------------------------
__device__ __forceinline__ v8f wmma_bf16(v16bf a, v16bf b, v8f c) {
  return __builtin_amdgcn_wmma_f32_16x16x32_bf16(
      /*neg_a=*/false, a, /*neg_b=*/false, b,
      /*c_mod=*/(short)0, c, /*reuse_a=*/false, /*reuse_b=*/false);
}

// Async 16-byte copy global -> LDS (GLOBAL_LOAD_ASYNC_TO_LDS_B128, ASYNCcnt).
// lds_off: LDS byte address (low 32 bits of generic shared pointer).
__device__ __forceinline__ void async_g2l_b128(unsigned lds_off, const void* gsrc) {
  asm volatile("global_load_async_to_lds_b128 %0, %1, off"
               :: "v"(lds_off), "v"((unsigned long long)(size_t)gsrc)
               : "memory");
}

__device__ __forceinline__ void wait_async0() {
#if defined(__has_builtin)
#if __has_builtin(__builtin_amdgcn_s_wait_asynccnt)
  __builtin_amdgcn_s_wait_asynccnt(0);
#else
  asm volatile("s_wait_asynccnt 0x0" ::: "memory");
#endif
#else
  asm volatile("s_wait_asynccnt 0x0" ::: "memory");
#endif
}

// A fragment: 16(M) x 32(K), source row-major [m][k] with leading dim ld.
__device__ __forceinline__ v16bf load_a_frag(const bf16* base, int ld) {
  int lane = threadIdx.x & 31;
  int half = lane >> 4, m = lane & 15;
  const bf16* rp = base + m * ld;
  v8bf lo = *(const v8bf*)(rp + 8 * half);
  v8bf hi = *(const v8bf*)(rp + 16 + 8 * half);
  v16bf a;
#pragma unroll
  for (int e = 0; e < 8; ++e) { a[e] = lo[e]; a[8 + e] = hi[e]; }
  return a;
}

// B fragment: 32(K) x 16(N), source stored TRANSPOSED as [n][k] (ld elems).
__device__ __forceinline__ v16bf load_b_frag(const bf16* bt, int ld) {
  int lane = threadIdx.x & 31;
  int half = lane >> 4, m = lane & 15;
  const bf16* rp = bt + m * ld;
  v8bf lo = *(const v8bf*)(rp + 8 * half);
  v8bf hi = *(const v8bf*)(rp + 16 + 8 * half);
  v16bf b;
#pragma unroll
  for (int e = 0; e < 8; ++e) { b[e] = lo[e]; b[8 + e] = hi[e]; }
  return b;
}

// ---------------------------------------------------------------------------
// Kernel 1: continuous relative position bias table (961 cells -> 6 heads)
// ---------------------------------------------------------------------------
__global__ void cpb_table_kernel(const float* __restrict__ scale,
                                 const float* __restrict__ w1,
                                 const float* __restrict__ b1,
                                 const float* __restrict__ w2,
                                 const float* __restrict__ b2,
                                 float* __restrict__ tbl) {
  int idx = blockIdx.x * blockDim.x + threadIdx.x;
  if (idx >= TBL) return;
  int i = idx / 31, j = idx % 31;
  float c0 = (float)(i - 15) * (8.0f / 15.0f);
  float c1 = (float)(j - 15) * (8.0f / 15.0f);
  float c2 = scale[0], c3 = scale[1];
  float acc[HEADS];
#pragma unroll
  for (int h = 0; h < HEADS; ++h) acc[h] = b2[h];
  for (int u = 0; u < 512; ++u) {
    float hv = c0 * w1[u * 4 + 0] + c1 * w1[u * 4 + 1] +
               c2 * w1[u * 4 + 2] + c3 * w1[u * 4 + 3] + b1[u];
    hv = fmaxf(hv, 0.0f);
#pragma unroll
    for (int h = 0; h < HEADS; ++h) acc[h] += hv * w2[h * 512 + u];
  }
#pragma unroll
  for (int h = 0; h < HEADS; ++h) tbl[idx * HEADS + h] = acc[h];
}

// ---------------------------------------------------------------------------
// Kernel 2: QKV GEMM (65536 x 576 x 192), bf16 WMMA, scatter into (b,h,n,d)
// (f32 -> bf16 conversion in flight, so staging stays on the VGPR path)
// ---------------------------------------------------------------------------
__global__ __launch_bounds__(256) void qkv_kernel(
    const float* __restrict__ x, const float* __restrict__ qkv_w,
    const float* __restrict__ qkv_b,
    bf16* __restrict__ qws, bf16* __restrict__ kws, bf16* __restrict__ vws) {
  __shared__ bf16 sA[128 * 32];
  __shared__ bf16 sB[64 * 32];   // weight rows (output cols) x K : already Bt
  int rowBase = blockIdx.x * 128;
  int colBase = blockIdx.y * 64;
  int tid = threadIdx.x;
  int waveId = tid >> 5, lane = tid & 31, half = lane >> 4, m = lane & 15;
  v8f acc[4] = {};

  for (int k0 = 0; k0 < DIM; k0 += 32) {
    int base = tid * 16;
#pragma unroll
    for (int i = 0; i < 16; ++i) {
      int idx = base + i;
      int r = idx >> 5, c = idx & 31;
      sA[idx] = (bf16)x[(size_t)(rowBase + r) * DIM + k0 + c];
    }
    if (tid < 128) {
#pragma unroll
      for (int i = 0; i < 16; ++i) {
        int idx = base + i;
        int r = idx >> 5, c = idx & 31;
        sB[idx] = (bf16)qkv_w[(size_t)(colBase + r) * DIM + k0 + c];
      }
    }
    __syncthreads();
    v16bf a = load_a_frag(sA + waveId * 16 * 32, 32);
#pragma unroll
    for (int nt = 0; nt < 4; ++nt) {
      v16bf b = load_b_frag(sB + nt * 16 * 32, 32);
      acc[nt] = wmma_bf16(a, b, acc[nt]);
    }
    __syncthreads();
  }

#pragma unroll
  for (int nt = 0; nt < 4; ++nt) {
#pragma unroll
    for (int r = 0; r < 8; ++r) {
      int gRow = rowBase + waveId * 16 + half * 8 + r;
      int c = colBase + nt * 16 + m;
      float val = acc[nt][r] + qkv_b[c];
      int which = c / DIM;
      int hh = (c % DIM) >> 5;
      int d = c & 31;
      int bw = gRow >> 8, n = gRow & 255;
      bf16* dst = (which == 0) ? qws : (which == 1) ? kws : vws;
      dst[((((size_t)bw * HEADS + hh) << 8) + n) * HD + d] = (bf16)val;
    }
  }
}

// ---------------------------------------------------------------------------
// Kernel 3: attention for one (b, h, 128-row q chunk).
// LDS: sQ 8K | sK 16K | sVt 16K | sBias ~4K | sS 128K  = 176000 B dynamic
// Q/K staged with GLOBAL_LOAD_ASYNC_TO_LDS_B128 (ASYNCcnt).
// ---------------------------------------------------------------------------
__global__ __launch_bounds__(256) void attn_kernel(
    const bf16* __restrict__ qws, const bf16* __restrict__ kws,
    const bf16* __restrict__ vws, const float* __restrict__ tbl,
    const float* __restrict__ mask, bf16* __restrict__ ows) {
  extern __shared__ char smem[];
  bf16*  sQ    = (bf16*)smem;                    // 128*32 bf16
  bf16*  sK    = (bf16*)(smem + 8192);           // 256*32 bf16  ([token][d] == Bt)
  bf16*  sVt   = (bf16*)(smem + 8192 + 16384);   // 32*256 bf16  ([d][token] == Bt)
  float* sBias = (float*)(smem + 40960);         // 961 f32 (padded)
  float* sS    = (float*)(smem + 44928);         // 128*256 f32

  int qhalf = blockIdx.x;          // 0..1
  int h     = blockIdx.y;          // 0..5
  int b     = blockIdx.z;          // 0..255
  int tid   = threadIdx.x;
  int qbase = qhalf * 128;
  size_t bh = ((size_t)b * HEADS + h) << 8;      // (b*H+h)*256

  unsigned ldsBase = (unsigned)(size_t)smem;     // LDS byte address of smem[0]

  // --- async stage Q (512 x 16B) and K (1024 x 16B), direct global -> LDS
  {
    const bf16* qsrc = qws + (bh + qbase) * HD;
#pragma unroll
    for (int c = tid, it = 0; it < 2; ++it, c += 256)
      async_g2l_b128(ldsBase + c * 16, qsrc + c * 8);
    const bf16* ksrc = kws + bh * HD;
#pragma unroll
    for (int c = tid, it = 0; it < 4; ++it, c += 256)
      async_g2l_b128(ldsBase + 8192 + c * 16, ksrc + c * 8);
  }
  // --- stage V transposed: sVt[d][token] (needs data reshuffle -> VGPR path)
  {
    const bf16* vr = vws + (bh + tid) * HD;
#pragma unroll
    for (int d = 0; d < HD; ++d) sVt[d * NT + tid] = vr[d];
  }
  for (int i = tid; i < TBL; i += 256) sBias[i] = tbl[i * HEADS + h];
  wait_async0();
  __syncthreads();

  int waveId = tid >> 5, lane = tid & 31, half = lane >> 4, m = lane & 15;
  const float qk_scale = 0.17677669529663687f;   // 1/sqrt(32)
  const float* mwin = mask + (size_t)(b & (NWMASK - 1)) * NT * NT;

  // ---- S = (Q * scale) K^T + bias + mask : one WMMA per 16x16 tile (K=hd=32)
  v16bf aq = load_a_frag(sQ + waveId * 16 * HD, HD);
#pragma unroll 4
  for (int ntile = 0; ntile < 16; ++ntile) {
    v16bf bk = load_b_frag(sK + ntile * 16 * HD, HD);
    v8f s = {};
    s = wmma_bf16(aq, bk, s);
#pragma unroll
    for (int r = 0; r < 8; ++r) {
      int rloc = waveId * 16 + half * 8 + r;
      int rowTok = qbase + rloc;
      int colTok = ntile * 16 + m;
      int ih = rowTok >> 4, iw = rowTok & 15;
      int jh = colTok >> 4, jw = colTok & 15;
      int ridx = (ih - jh + 15) * 31 + (iw - jw + 15);
      sS[rloc * NT + colTok] = s[r] * qk_scale + sBias[ridx] +
                               mwin[(size_t)rowTok * NT + colTok];
    }
  }
  __syncthreads();

  // ---- row softmax (one thread per row)
  if (tid < 128) {
    float* row = sS + tid * NT;
    float mx = -1e30f;
    for (int j = 0; j < NT; ++j) mx = fmaxf(mx, row[j]);
    float sum = 0.0f;
    for (int j = 0; j < NT; ++j) { float e = __expf(row[j] - mx); row[j] = e; sum += e; }
    float inv = 1.0f / sum;
    for (int j = 0; j < NT; ++j) row[j] *= inv;
  }
  __syncthreads();

  // ---- O = P V : K loop over 256 tokens, 2 N-tiles of hd
  v8f o0 = {}, o1 = {};
  for (int k0 = 0; k0 < NT; k0 += 32) {
    const float* rp = sS + (waveId * 16 + m) * NT + k0;
    v16bf pa;
#pragma unroll
    for (int e = 0; e < 8; ++e) {
      pa[e]     = (bf16)rp[8 * half + e];
      pa[8 + e] = (bf16)rp[16 + 8 * half + e];
    }
    v16bf b0 = load_b_frag(sVt + 0 * 16 * NT + k0, NT);
    v16bf b1 = load_b_frag(sVt + 1 * 16 * NT + k0, NT);
    o0 = wmma_bf16(pa, b0, o0);
    o1 = wmma_bf16(pa, b1, o1);
  }
#pragma unroll
  for (int r = 0; r < 8; ++r) {
    int tok = qbase + waveId * 16 + half * 8 + r;
    size_t orow = ((size_t)b * NT + tok) * DIM + (size_t)h * HD;
    ows[orow + m]      = (bf16)o0[r];
    ows[orow + 16 + m] = (bf16)o1[r];
  }
}

// ---------------------------------------------------------------------------
// Kernel 4: proj GEMM (65536 x 192 x 192) + bias -> f32 output
// A is already bf16 -> async global->LDS staging for the A tile.
// ---------------------------------------------------------------------------
__global__ __launch_bounds__(256) void proj_kernel(
    const bf16* __restrict__ ows, const float* __restrict__ proj_w,
    const float* __restrict__ proj_b, float* __restrict__ out) {
  __shared__ bf16 sA[128 * 32];
  __shared__ bf16 sB[64 * 32];
  int rowBase = blockIdx.x * 128;
  int colBase = blockIdx.y * 64;
  int tid = threadIdx.x;
  int waveId = tid >> 5, lane = tid & 31, half = lane >> 4, m = lane & 15;
  unsigned ldsA = (unsigned)(size_t)&sA[0];
  v8f acc[4] = {};

  for (int k0 = 0; k0 < DIM; k0 += 32) {
    // A tile: 128 rows x 64B -> 512 async B128 transfers (2 per thread)
#pragma unroll
    for (int c = tid, it = 0; it < 2; ++it, c += 256) {
      int r = c >> 2, part = c & 3;
      async_g2l_b128(ldsA + c * 16,
                     ows + (size_t)(rowBase + r) * DIM + k0 + part * 8);
    }
    if (tid < 128) {
      int base = tid * 16;
#pragma unroll
      for (int i = 0; i < 16; ++i) {
        int idx = base + i;
        int r = idx >> 5, c = idx & 31;
        sB[idx] = (bf16)proj_w[(size_t)(colBase + r) * DIM + k0 + c];
      }
    }
    wait_async0();
    __syncthreads();
    v16bf a = load_a_frag(sA + waveId * 16 * 32, 32);
#pragma unroll
    for (int nt = 0; nt < 4; ++nt) {
      v16bf bb = load_b_frag(sB + nt * 16 * 32, 32);
      acc[nt] = wmma_bf16(a, bb, acc[nt]);
    }
    __syncthreads();
  }

#pragma unroll
  for (int nt = 0; nt < 4; ++nt) {
#pragma unroll
    for (int r = 0; r < 8; ++r) {
      int gRow = rowBase + waveId * 16 + half * 8 + r;
      int c = colBase + nt * 16 + m;
      out[(size_t)gRow * DIM + c] = acc[nt][r] + proj_b[c];
    }
  }
}

// ---------------------------------------------------------------------------
extern "C" void kernel_launch(void* const* d_in, const int* in_sizes, int n_in,
                              void* d_out, int out_size, void* d_ws, size_t ws_size,
                              hipStream_t stream) {
  const float* x      = (const float*)d_in[0];
  const float* scale  = (const float*)d_in[1];
  const float* mask   = (const float*)d_in[2];
  const float* qkv_w  = (const float*)d_in[3];
  const float* qkv_b  = (const float*)d_in[4];
  const float* cpb_w1 = (const float*)d_in[5];
  const float* cpb_b1 = (const float*)d_in[6];
  const float* cpb_w2 = (const float*)d_in[7];
  const float* cpb_b2 = (const float*)d_in[8];
  const float* proj_w = (const float*)d_in[9];
  const float* proj_b = (const float*)d_in[10];

  // workspace layout
  const size_t QKV_BYTES = (size_t)BWIN * HEADS * NT * HD * sizeof(bf16); // 25165824
  char* ws = (char*)d_ws;
  float* tbl = (float*)ws;
  size_t off = 23296;                       // 961*6*4 = 23064, padded to 256B
  bf16* qws = (bf16*)(ws + off); off += QKV_BYTES;
  bf16* kws = (bf16*)(ws + off); off += QKV_BYTES;
  bf16* vws = (bf16*)(ws + off); off += QKV_BYTES;
  bf16* ows = (bf16*)(ws + off);            // (b, n, C) bf16

  cpb_table_kernel<<<4, 256, 0, stream>>>(scale, cpb_w1, cpb_b1, cpb_w2, cpb_b2, tbl);
  qkv_kernel<<<dim3(512, 9), 256, 0, stream>>>(x, qkv_w, qkv_b, qws, kws, vws);
  attn_kernel<<<dim3(2, HEADS, BWIN), 256, 176000, stream>>>(qws, kws, vws, tbl, mask, ows);
  proj_kernel<<<dim3(512, 3), 256, 0, stream>>>(ows, proj_w, proj_b, (float*)d_out);
}
